// MultiHeadAttentionBlock_57930518888919
// MI455X (gfx1250) — matrix operand order repaired and checked by reading, not compiled, not run on hardware
//
#include <hip/hip_runtime.h>

// ---- types matching gfx1250 WMMA builtins ----
typedef __bf16 bf16_t;
typedef __attribute__((ext_vector_type(16))) __bf16 v16bf;
typedef __attribute__((ext_vector_type(8)))  __bf16 v8bf;
typedef __attribute__((ext_vector_type(8)))  float  v8f;
typedef unsigned int u32;
typedef __attribute__((ext_vector_type(4))) u32 u32x4;
typedef __attribute__((ext_vector_type(8))) u32 u32x8;

#define N_TOK 4096
#define DIM   2048
#define HEADS 16
#define DK    128

#if __has_builtin(__builtin_amdgcn_sched_barrier)
#define SCHED_FENCE() __builtin_amdgcn_sched_barrier(0)
#else
#define SCHED_FENCE()
#endif

// ---- fragment loaders (contiguous k order k = 16h+s, same for A and B) ----
// A 16x32 bf16: lane l holds row M=l%16; low lanes K=0..15, high lanes K=16..31.
// Caller passes p = row_base + (hi?16:0); we read p[0..7] and p[8..15].
__device__ __forceinline__ v16bf load_a_frag(const bf16_t* p) {
  v8bf lo = *(const v8bf*)(p);
  v8bf hh = *(const v8bf*)(p + 8);
  v16bf r;
#pragma unroll
  for (int i = 0; i < 8; ++i) { r[i] = lo[i]; r[i + 8] = hh[i]; }
  return r;
}
// B 32x16 bf16: lane l holds column n=l%16, 16 contiguous K values (low lanes K=0..15, high K=16..31).
__device__ __forceinline__ v16bf load_b_frag(const bf16_t* p) {
  return *(const v16bf*)p;
}

// ---- CDNA5 async copy: global -> LDS, 16 bytes per lane, tracked by ASYNCcnt ----
__device__ __forceinline__ void async_copy_b128(unsigned lds_off, const bf16_t* gsrc) {
  asm volatile("global_load_async_to_lds_b128 %0, %1, off"
               :: "v"(lds_off), "v"((unsigned long long)(size_t)gsrc)
               : "memory");
}

// ---- CDNA5 Tensor Data Mover: one 2D tile global -> LDS, tracked by TENSORcnt ----
// D# group0: [1:0]=count=1, [63:32]=lds_addr, [120:64]=global_addr, [127:126]=type=2
__device__ __forceinline__ u32x4 tdm_g0(unsigned lds_addr, const void* gptr) {
  const unsigned long long ga = (unsigned long long)(size_t)gptr;
  u32x4 g;
  g[0] = 1u;
  g[1] = lds_addr;
  g[2] = (u32)ga;
  g[3] = ((u32)(ga >> 32) & 0x01FFFFFFu) | 0x80000000u;
  return g;
}
// D# group1: data_size=2B; 2D tensor dim0 x dim1, tile dim0 x dim1, dim0 stride (elements)
__device__ __forceinline__ u32x8 tdm_g1(u32 dim0, u32 dim1, u32 tile0, u32 tile1,
                                        u32 stride0) {
  u32x8 g;
  g[0] = 1u << 16;                                   // data_size = 2 bytes
  g[1] = (dim0 & 0xFFFFu) << 16;                     // tensor_dim0[15:0]
  g[2] = (dim0 >> 16) | ((dim1 & 0xFFFFu) << 16);    // dim0[31:16] | dim1[15:0]
  g[3] = (dim1 >> 16) | (tile0 << 16);               // dim1[31:16] | tile_dim0
  g[4] = tile1;                                      // tile_dim1 (tile_dim2=0)
  g[5] = stride0;                                    // tensor_dim0_stride[31:0]
  g[6] = 0;
  g[7] = 0;
  return g;
}
__device__ __forceinline__ void tdm_load_to_lds(u32x4 g0, u32x8 g1) {
  asm volatile("tensor_load_to_lds %0, %1" :: "s"(g0), "s"(g1) : "memory");
}

// ---- fp32 -> bf16 conversion ----
__global__ void cvt_bf16_kernel(const float* __restrict__ src,
                                bf16_t* __restrict__ dst, int n) {
  int i = blockIdx.x * blockDim.x + threadIdx.x;
  int stride = gridDim.x * blockDim.x;
  for (; i < n; i += stride) dst[i] = (bf16_t)src[i];
}

// ---- GEMM: out = (A @ Bt^T + bias) * scale ----
// MODE 0: bf16 out [M,Ncol]; MODE 1: bf16 out transposed [Ncol,M]; MODE 2: fp32 out [M,Ncol].
template <int MODE>
__global__ void __launch_bounds__(128) gemm_wmma_kernel(
    const bf16_t* __restrict__ A, const bf16_t* __restrict__ Bt,
    const float* __restrict__ bias, void* __restrict__ outv,
    float scale, int M, int K, int Ncol) {
  const int lane = threadIdx.x & 31;
  const int wave = threadIdx.x >> 5;
  const int hi   = lane >> 4;
  const int ln   = lane & 15;
  const int n0 = blockIdx.y * 64 + wave * 16;
  const int j0 = blockIdx.x * 64;

  v8f acc[4];
#pragma unroll
  for (int t = 0; t < 4; ++t) acc[t] = (v8f){};

  const bf16_t* arow = A + (size_t)(n0 + ln) * K + (hi ? 16 : 0);
  const bf16_t* brow[4];
#pragma unroll
  for (int t = 0; t < 4; ++t)
    brow[t] = Bt + (size_t)(j0 + t * 16 + ln) * K + (hi ? 16 : 0);

  v16bf ac = load_a_frag(arow);
  v16bf bc[4];
#pragma unroll
  for (int t = 0; t < 4; ++t) bc[t] = load_b_frag(brow[t]);

  for (int kc = 0; kc < K - 32; kc += 32) {
    v16bf an = load_a_frag(arow + kc + 32);
    v16bf bn[4];
#pragma unroll
    for (int t = 0; t < 4; ++t) bn[t] = load_b_frag(brow[t] + kc + 32);
    SCHED_FENCE();  // next-chunk loads issue before current WMMAs
#pragma unroll
    for (int t = 0; t < 4; ++t)
      acc[t] = __builtin_amdgcn_wmma_f32_16x16x32_bf16(
          false, ac, false, bc[t], (short)0, acc[t], false, false);
    ac = an;
#pragma unroll
    for (int t = 0; t < 4; ++t) bc[t] = bn[t];
  }
#pragma unroll
  for (int t = 0; t < 4; ++t)
    acc[t] = __builtin_amdgcn_wmma_f32_16x16x32_bf16(
        false, ac, false, bc[t], (short)0, acc[t], false, false);

#pragma unroll
  for (int t = 0; t < 4; ++t) {
    const int j = j0 + t * 16 + ln;
    const float bj = bias[j];
#pragma unroll
    for (int r = 0; r < 8; ++r) {
      const int n = n0 + r + hi * 8;
      const float v = (acc[t][r] + bj) * scale;
      if (MODE == 0)      ((bf16_t*)outv)[(size_t)n * Ncol + j] = (bf16_t)v;
      else if (MODE == 1) ((bf16_t*)outv)[(size_t)j * M + n]    = (bf16_t)v;
      else                ((float*)outv)[(size_t)n * Ncol + j]  = v;
    }
  }
}

// ---- flash attention ----
// Block = 4 waves = head h, 64 query rows. 32-key K/V tiles staged in LDS,
// double-buffered: K tile via one TDM tensor_load_to_lds (wave 0, TENSORcnt),
// V tile via per-lane async global->LDS copies (ASYNCcnt).
__global__ void __launch_bounds__(128) flash_attn_kernel(
    const bf16_t* __restrict__ Qp, const bf16_t* __restrict__ Kp,
    const bf16_t* __restrict__ Vt, bf16_t* __restrict__ Xc) {
  __shared__ __align__(16) bf16_t kbuf[2][32 * DK];   // [key][dim]
  __shared__ __align__(16) bf16_t vbuf[2][DK * 32];   // [dim][key]
  __shared__ __align__(16) bf16_t pbuf[4][16 * 32];   // per-wave P bounce

  const int tid  = threadIdx.x;
  const int lane = tid & 31;
  const int wave = tid >> 5;
  const int hi   = lane >> 4;
  const int ln   = lane & 15;
  const int h  = blockIdx.x >> 6;
  const int n0 = (blockIdx.x & 63) * 64 + wave * 16;

  const bf16_t* kg = Kp + h * DK;                  // K tile base + j0*DIM
  const bf16_t* vg = Vt + (size_t)h * DK * N_TOK;  // V tile base + j0

  // K tile D# group1: 128x32 tile of a [*,2048]-strided tensor (bf16)
  const u32x8 g1k = tdm_g1(/*dim0=*/DK, /*dim1=*/32, /*tile0=*/DK, /*tile1=*/32,
                           /*stride0=*/DIM);

  // wave 0: one TDM op per K tile
  auto issue_k = [&](int buf, int j0) {
    tdm_load_to_lds(tdm_g0((unsigned)(size_t)&kbuf[buf][0],
                           kg + (size_t)j0 * DIM), g1k);
  };
  // all lanes: 512 x 16B async chunks for the V tile
  auto issue_v = [&](int buf, int j0) {
    const unsigned vb_off = (unsigned)(size_t)&vbuf[buf][0];
#pragma unroll
    for (int i = 0; i < 4; ++i) {
      const int c = tid + i * 128;
      async_copy_b128(vb_off + c * 16,
                      vg + (size_t)(c >> 2) * N_TOK + j0 + (c & 3) * 8);
    }
  };

  // Q A-fragments, 4 K-chunks of 32 covering dk=128
  const bf16_t* qrow = Qp + (size_t)(n0 + ln) * DIM + h * DK + (hi ? 16 : 0);
  v16bf aq[4];
#pragma unroll
  for (int c = 0; c < 4; ++c) aq[c] = load_a_frag(qrow + c * 32);

  v8f o[8];
#pragma unroll
  for (int t = 0; t < 8; ++t) o[t] = (v8f){};
  float m[8], l[8];
#pragma unroll
  for (int r = 0; r < 8; ++r) { m[r] = -3.0e38f; l[r] = 0.0f; }
  bf16_t* pb = &pbuf[wave][0];

  if (wave == 0) issue_k(0, 0);
  issue_v(0, 0);
  for (int j0 = 0; j0 < N_TOK; j0 += 32) {
    const int cur = (j0 >> 5) & 1;
    const bool has_next = (j0 + 32 < N_TOK);
    if (has_next) {
      if (wave == 0) issue_k(cur ^ 1, j0 + 32);
      issue_v(cur ^ 1, j0 + 32);
      asm volatile("s_wait_asynccnt 0x4" ::: "memory");   // prev V tile done
      if (wave == 0) __builtin_amdgcn_s_wait_tensorcnt(1);  // prev K tile done
    } else {
      asm volatile("s_wait_asynccnt 0x0" ::: "memory");
      if (wave == 0) __builtin_amdgcn_s_wait_tensorcnt(0);
    }
    __syncthreads();  // staged tiles visible to all waves
    const bf16_t* kb = &kbuf[cur][0];
    const bf16_t* vb = &vbuf[cur][0];

    // ---- S = Q K^T : two 16x16 tiles over 32 keys ----
    v16bf bk[8];
#pragma unroll
    for (int c = 0; c < 4; ++c) {
      bk[c]     = load_b_frag(kb + (size_t)ln * DK + c * 32 + (hi ? 16 : 0));
      bk[4 + c] = load_b_frag(kb + (size_t)(16 + ln) * DK + c * 32 + (hi ? 16 : 0));
    }
    SCHED_FENCE();  // all ds_loads issue, then WMMAs run back-to-back
    v8f s0 = (v8f){}, s1 = (v8f){};
#pragma unroll
    for (int c = 0; c < 4; ++c) {
      s0 = __builtin_amdgcn_wmma_f32_16x16x32_bf16(
          false, aq[c], false, bk[c], (short)0, s0, false, false);
      s1 = __builtin_amdgcn_wmma_f32_16x16x32_bf16(
          false, aq[c], false, bk[4 + c], (short)0, s1, false, false);
    }

    // ---- online softmax ----
#pragma unroll
    for (int r = 0; r < 8; ++r) {
      float rm = fmaxf(s0[r], s1[r]);
      rm = fmaxf(rm, __shfl_xor(rm, 1, 32));
      rm = fmaxf(rm, __shfl_xor(rm, 2, 32));
      rm = fmaxf(rm, __shfl_xor(rm, 4, 32));
      rm = fmaxf(rm, __shfl_xor(rm, 8, 32));
      const float mn  = fmaxf(m[r], rm);
      const float scl = __expf(m[r] - mn);
      m[r] = mn;
      const float p0 = __expf(s0[r] - mn);
      const float p1 = __expf(s1[r] - mn);
      float rs = p0 + p1;
      rs += __shfl_xor(rs, 1, 32);
      rs += __shfl_xor(rs, 2, 32);
      rs += __shfl_xor(rs, 4, 32);
      rs += __shfl_xor(rs, 8, 32);
      l[r] = l[r] * scl + rs;
      const int row = r + hi * 8;
      pb[row * 32 + ln]      = (bf16_t)p0;
      pb[row * 32 + 16 + ln] = (bf16_t)p1;
#pragma unroll
      for (int t = 0; t < 8; ++t) o[t][r] *= scl;
    }
    asm volatile("s_wait_dscnt 0x0" ::: "memory");
    v16bf ap = load_a_frag(pb + ln * 32 + (hi ? 16 : 0));
    asm volatile("" ::: "memory");

    // ---- O += P @ V ----
    v16bf bv[8];
#pragma unroll
    for (int t = 0; t < 8; ++t)
      bv[t] = load_b_frag(vb + (size_t)(t * 16 + ln) * 32 + (hi ? 16 : 0));
    SCHED_FENCE();
#pragma unroll
    for (int t = 0; t < 8; ++t)
      o[t] = __builtin_amdgcn_wmma_f32_16x16x32_bf16(
          false, ap, false, bv[t], (short)0, o[t], false, false);

    __syncthreads();  // readers done before buffer is refilled
  }

  // ---- epilogue: divide by l, faithful [H,N,dk]->[N,H*dk] reshape ----
#pragma unroll
  for (int r = 0; r < 8; ++r) {
    const float inv = 1.0f / l[r];
    const int n = n0 + r + hi * 8;
    const int row_x = h * 256 + (n >> 4);
    const int col_b = (n & 15) * 128;
#pragma unroll
    for (int t = 0; t < 8; ++t)
      Xc[(size_t)row_x * DIM + col_b + t * 16 + ln] = (bf16_t)(o[t][r] * inv);
  }
}

extern "C" void kernel_launch(void* const* d_in, const int* in_sizes, int n_in,
                              void* d_out, int out_size, void* d_ws, size_t ws_size,
                              hipStream_t stream) {
  const float* q  = (const float*)d_in[0];
  const float* k  = (const float*)d_in[1];
  const float* v  = (const float*)d_in[2];
  const float* Wq = (const float*)d_in[3];
  const float* bq = (const float*)d_in[4];
  const float* Wk = (const float*)d_in[5];
  const float* bk = (const float*)d_in[6];
  const float* Wv = (const float*)d_in[7];
  const float* bv = (const float*)d_in[8];
  const float* Wo = (const float*)d_in[9];
  const float* bo = (const float*)d_in[10];
  float* out = (float*)d_out;

  char* ws = (char*)d_ws;
  const size_t SZ_ND = (size_t)N_TOK * DIM * sizeof(bf16_t);
  const size_t SZ_DD = (size_t)DIM * DIM * sizeof(bf16_t);
  bf16_t* qb  = (bf16_t*)ws; ws += SZ_ND;
  bf16_t* kb  = (bf16_t*)ws; ws += SZ_ND;
  bf16_t* vb  = (bf16_t*)ws; ws += SZ_ND;
  bf16_t* Wqb = (bf16_t*)ws; ws += SZ_DD;
  bf16_t* Wkb = (bf16_t*)ws; ws += SZ_DD;
  bf16_t* Wvb = (bf16_t*)ws; ws += SZ_DD;
  bf16_t* Wob = (bf16_t*)ws; ws += SZ_DD;
  bf16_t* Qp  = (bf16_t*)ws; ws += SZ_ND;
  bf16_t* Kp  = (bf16_t*)ws; ws += SZ_ND;
  bf16_t* Vt  = (bf16_t*)ws; ws += SZ_ND;
  bf16_t* Xc  = (bf16_t*)ws; ws += SZ_ND;

  const int ND = N_TOK * DIM, DD = DIM * DIM;
  cvt_bf16_kernel<<<2048, 256, 0, stream>>>(q,  qb,  ND);
  cvt_bf16_kernel<<<2048, 256, 0, stream>>>(k,  kb,  ND);
  cvt_bf16_kernel<<<2048, 256, 0, stream>>>(v,  vb,  ND);
  cvt_bf16_kernel<<<2048, 256, 0, stream>>>(Wq, Wqb, DD);
  cvt_bf16_kernel<<<2048, 256, 0, stream>>>(Wk, Wkb, DD);
  cvt_bf16_kernel<<<2048, 256, 0, stream>>>(Wv, Wvb, DD);
  cvt_bf16_kernel<<<2048, 256, 0, stream>>>(Wo, Wob, DD);

  const dim3 ggrid(DIM / 64, N_TOK / 64);
  const float qscale = 0.08838834764831845f;  // 1/sqrt(128), folded into Q
  gemm_wmma_kernel<0><<<ggrid, 128, 0, stream>>>(qb, Wqb, bq, Qp, qscale, N_TOK, DIM, DIM);
  gemm_wmma_kernel<0><<<ggrid, 128, 0, stream>>>(kb, Wkb, bk, Kp, 1.0f,   N_TOK, DIM, DIM);
  gemm_wmma_kernel<1><<<ggrid, 128, 0, stream>>>(vb, Wvb, bv, Vt, 1.0f,   N_TOK, DIM, DIM);

  flash_attn_kernel<<<HEADS * (N_TOK / 64), 128, 0, stream>>>(Qp, Kp, Vt, Xc);

  gemm_wmma_kernel<2><<<ggrid, 128, 0, stream>>>(Xc, Wob, bo, out, 1.0f, N_TOK, DIM, DIM);
}
